// LGMSA_71614284693736
// MI455X (gfx1250) — matrix operand order repaired and checked by reading, not compiled, not run on hardware
//
#include <hip/hip_runtime.h>
#include <hip/hip_bf16.h>

#define SEQ     2048
#define NHEADS  16
#define HD      64
#define DIM     1024
#define NBATCH  2
#define WSZ     64
#define MROWS   (NBATCH * SEQ)      /* 4096 */
#define SCALEF  0.125f              /* HEAD_DIM^-0.5 */

typedef __attribute__((ext_vector_type(16))) __bf16 bf16x16;
typedef __attribute__((ext_vector_type(8)))  __bf16 bf16x8;
typedef __attribute__((ext_vector_type(8)))  float  floatx8;
typedef int v4i __attribute__((vector_size(16)));

// ---------------------------------------------------------------------------
// CDNA5 async global->LDS staging (ASYNCcnt path), with safe fallback.
// Builtin signature (probe-confirmed by diagnostic):
//   void(v4i addrspace(1)* src, v4i addrspace(3)* dst, Ii offset, Ii cpol)
// ---------------------------------------------------------------------------
#if defined(__gfx1250__) && __has_builtin(__builtin_amdgcn_global_load_async_to_lds_b128)
#define HAVE_ASYNC 1
#else
#define HAVE_ASYNC 0
#endif

__device__ __forceinline__ void cp_g2l_b128(const __bf16* g, __bf16* l) {
#if HAVE_ASYNC
  __builtin_amdgcn_global_load_async_to_lds_b128(
      (__attribute__((address_space(1))) v4i*)(__bf16*)g,
      (__attribute__((address_space(3))) v4i*)l, 0, 0);
#else
  *(bf16x8*)l = *(const bf16x8*)g;
#endif
}
__device__ __forceinline__ void wait_async() {
#if HAVE_ASYNC
#if __has_builtin(__builtin_amdgcn_s_wait_asynccnt)
  __builtin_amdgcn_s_wait_asynccnt(0);
#else
  asm volatile("s_wait_asynccnt 0x0" ::: "memory");
#endif
#endif
}

// ---------------------------------------------------------------------------
// WMMA fragment loaders (wave32, v_wmma_f32_16x16x32_bf16).
// A 16x32 bf16: lane%16 = row; half g elems 0..7 = K+8g.., 8..15 = K+16+8g..
// B 32x16 bf16 (LDS stores B transposed as [n][k]): lane%16 = col; half g
//   elems 0..7 = K+8g.., 8..15 = K+16+8g.. (same as A).
// C/D 16x16 f32: elem v -> row v + 8g, col = lane%16.
// ---------------------------------------------------------------------------
__device__ __forceinline__ bf16x16 frag_a(const __bf16* rowbase, int g) {
  bf16x8 lo = *(const bf16x8*)(rowbase + (g << 3));
  bf16x8 hi = *(const bf16x8*)(rowbase + 16 + (g << 3));
  return __builtin_shufflevector(lo, hi, 0,1,2,3,4,5,6,7,8,9,10,11,12,13,14,15);
}
__device__ __forceinline__ bf16x16 frag_b(const __bf16* rowbase, int g) {
  bf16x8 lo = *(const bf16x8*)(rowbase + (g << 3));
  bf16x8 hi = *(const bf16x8*)(rowbase + 16 + (g << 3));
  return __builtin_shufflevector(lo, hi, 0,1,2,3,4,5,6,7,8,9,10,11,12,13,14,15);
}
#define WMMA_BF16(A, B, C) \
  __builtin_amdgcn_wmma_f32_16x16x32_bf16(false, (A), false, (B), (short)0, (C), false, false)

// ---------------------------------------------------------------------------
// elementwise f32 -> bf16
// ---------------------------------------------------------------------------
__global__ void cvt_f32_bf16(const float* __restrict__ src, __bf16* __restrict__ dst, int n) {
  int i = blockIdx.x * blockDim.x + threadIdx.x;
  if (i < n) dst[i] = (__bf16)src[i];
}

// ---------------------------------------------------------------------------
// tiled transpose + convert: src f32 [K][N] -> dst bf16 [N][K]
// ---------------------------------------------------------------------------
__global__ __launch_bounds__(256)
void cvtT_f32_bf16(const float* __restrict__ src, __bf16* __restrict__ dst,
                   int Kd, int Nd) {
  __shared__ float tile[32][33];
  const int kb = blockIdx.y * 32, nb = blockIdx.x * 32;
  const int tx = threadIdx.x & 31, ty = threadIdx.x >> 5;   // ty: 0..7
#pragma unroll
  for (int i = ty; i < 32; i += 8)
    tile[i][tx] = src[(size_t)(kb + i) * Nd + nb + tx];
  __syncthreads();
#pragma unroll
  for (int i = ty; i < 32; i += 8)
    dst[(size_t)(nb + i) * Kd + kb + tx] = (__bf16)tile[tx][i];
}

// ---------------------------------------------------------------------------
// bf16 GEMM: C[M,N] = A[M,K] * B[K,N] + bias[N], B given pre-transposed BT[N][K].
// 256 threads = 8 waves; block tile 128x128; each wave 32x64 (8 WMMA / K-step).
// Optional outputs: Cb (bf16 row-major), Cf (f32 row-major), CbT (bf16 col-major).
// ---------------------------------------------------------------------------
__global__ __launch_bounds__(256)
void gemm_bf16(const __bf16* __restrict__ A, int lda,
               const __bf16* __restrict__ BT, int ldbt,
               const float*  __restrict__ bias,
               __bf16* __restrict__ Cb, float* __restrict__ Cf,
               __bf16* __restrict__ CbT, int ldc, int ldct, int K) {
  __shared__ __bf16 As[128][32];
  __shared__ __bf16 Bs[128][32];     // Bs[n][k]
  const int tid = threadIdx.x;
  const int wave = tid >> 5, lane = tid & 31, g = lane >> 4, ln = lane & 15;
  const int mw = wave >> 1, nw = wave & 1;          // 4x2 wave grid
  const int m0 = blockIdx.y * 128, n0 = blockIdx.x * 128;

  floatx8 acc[2][4] = {};
  for (int kb = 0; kb < K; kb += 32) {
    __syncthreads();                                 // LDS reuse guard
    {
      int r = tid >> 1, seg = (tid & 1) << 4;
      const __bf16* asrc = A  + (size_t)(m0 + r) * lda  + kb + seg;
      const __bf16* bsrc = BT + (size_t)(n0 + r) * ldbt + kb + seg;
      cp_g2l_b128(asrc,     &As[r][seg]);
      cp_g2l_b128(asrc + 8, &As[r][seg + 8]);
      cp_g2l_b128(bsrc,     &Bs[r][seg]);
      cp_g2l_b128(bsrc + 8, &Bs[r][seg + 8]);
      if (kb + 32 < K) {
        __builtin_prefetch(asrc + 32, 0, 1);
        __builtin_prefetch(bsrc + 32, 0, 1);
      }
    }
    wait_async();
    __syncthreads();
    bf16x16 a0 = frag_a(&As[mw * 32 + ln][0], g);
    bf16x16 a1 = frag_a(&As[mw * 32 + 16 + ln][0], g);
#pragma unroll
    for (int nt = 0; nt < 4; ++nt) {
      bf16x16 b = frag_b(&Bs[nw * 64 + nt * 16 + ln][0], g);
      acc[0][nt] = WMMA_BF16(a0, b, acc[0][nt]);
      acc[1][nt] = WMMA_BF16(a1, b, acc[1][nt]);
    }
  }
#pragma unroll
  for (int at = 0; at < 2; ++at)
#pragma unroll
    for (int nt = 0; nt < 4; ++nt) {
      int col = n0 + nw * 64 + nt * 16 + ln;
      float bv = bias ? bias[col] : 0.f;
#pragma unroll
      for (int v = 0; v < 8; ++v) {
        int row = m0 + mw * 32 + at * 16 + v + 8 * g;
        float val = acc[at][nt][v] + bv;
        if (Cb)  Cb[(size_t)row * ldc + col]  = (__bf16)val;
        if (Cf)  Cf[(size_t)row * ldc + col]  = val;
        if (CbT) CbT[(size_t)col * ldct + row] = (__bf16)val;  // V^T layout
      }
    }
}

// ---------------------------------------------------------------------------
// Windowed attention: one block per (batch, head, window); 4 waves x 16 rows.
// S = Q K^T * scale + bias[j]; softmax; O = P V. V comes pre-transposed.
// ---------------------------------------------------------------------------
__global__ __launch_bounds__(128)
void windowed_attn(const __bf16* __restrict__ wqk,   // [4096][2048] (q|k)
                   const __bf16* __restrict__ vT,    // [1024][4096]
                   const float*  __restrict__ wpos,  // [2*WSZ-1]
                   float* __restrict__ outf) {       // [4096][1024]
  __shared__ __bf16 qs[64][64], ks[64][64], vts[64][64], ps[64][64];
  __shared__ float sbias[64];
  const int bx = blockIdx.x;
  const int win = bx & 31, head = (bx >> 5) & 15, batch = bx >> 9;
  const int tid = threadIdx.x;
  const int wave = tid >> 5, lane = tid & 31, g = lane >> 4, ln = lane & 15;
  const int tbase = batch * SEQ + win * 64;

  {
    int r = tid >> 1, sb = (tid & 1) * 32;
    const __bf16* qsrc = wqk + (size_t)(tbase + r) * (2 * DIM) + head * HD + sb;
    const __bf16* ksrc = qsrc + DIM;
    const __bf16* vsrc = vT + (size_t)(head * HD + r) * MROWS + tbase + sb;
#pragma unroll
    for (int s = 0; s < 4; ++s) {
      cp_g2l_b128(qsrc + s * 8, &qs[r][sb + s * 8]);
      cp_g2l_b128(ksrc + s * 8, &ks[r][sb + s * 8]);
      cp_g2l_b128(vsrc + s * 8, &vts[r][sb + s * 8]);
    }
  }
  if (tid < 64) sbias[tid] = wpos[(WSZ - 1) - tid];
  wait_async();
  __syncthreads();

  const int i0 = wave * 16;
  floatx8 sc[4] = {};
#pragma unroll
  for (int d0 = 0; d0 < 64; d0 += 32) {
    bf16x16 a = frag_a(&qs[i0 + ln][d0], g);
#pragma unroll
    for (int nt = 0; nt < 4; ++nt) {
      bf16x16 b = frag_b(&ks[nt * 16 + ln][d0], g);
      sc[nt] = WMMA_BF16(a, b, sc[nt]);
    }
  }
  float pv[4][8];
#pragma unroll
  for (int nt = 0; nt < 4; ++nt)
#pragma unroll
    for (int v = 0; v < 8; ++v)
      pv[nt][v] = sc[nt][v] * SCALEF + sbias[nt * 16 + ln];

#pragma unroll
  for (int v = 0; v < 8; ++v) {          // softmax over 64 keys of row v+8g
    float m = fmaxf(fmaxf(pv[0][v], pv[1][v]), fmaxf(pv[2][v], pv[3][v]));
    for (int off = 8; off; off >>= 1) m = fmaxf(m, __shfl_xor(m, off, 32));
    float sum = 0.f;
#pragma unroll
    for (int nt = 0; nt < 4; ++nt) { pv[nt][v] = __expf(pv[nt][v] - m); sum += pv[nt][v]; }
    for (int off = 8; off; off >>= 1) sum += __shfl_xor(sum, off, 32);
    float inv = 1.f / sum;
    int row = i0 + v + 8 * g;
#pragma unroll
    for (int nt = 0; nt < 4; ++nt) ps[row][nt * 16 + ln] = (__bf16)(pv[nt][v] * inv);
  }
  __syncthreads();

  floatx8 oc[4] = {};
#pragma unroll
  for (int j0 = 0; j0 < 64; j0 += 32) {
    bf16x16 a = frag_a(&ps[i0 + ln][j0], g);
#pragma unroll
    for (int dt = 0; dt < 4; ++dt) {
      bf16x16 b = frag_b(&vts[dt * 16 + ln][j0], g);
      oc[dt] = WMMA_BF16(a, b, oc[dt]);
    }
  }
#pragma unroll
  for (int dt = 0; dt < 4; ++dt) {
    int col = head * HD + dt * 16 + ln;
#pragma unroll
    for (int v = 0; v < 8; ++v) {
      int row = tbase + i0 + v + 8 * g;
      outf[(size_t)row * DIM + col] = oc[dt][v];
    }
  }
}

// ---------------------------------------------------------------------------
// Global attention (flash-style): one block per (batch, head, 64-row q block),
// online softmax over 32 key blocks; fuses windowed+global add, emits bf16 A.
// ---------------------------------------------------------------------------
__global__ __launch_bounds__(128)
void global_attn(const __bf16* __restrict__ qk,     // [4096][2048] (q|k)
                 const __bf16* __restrict__ vT,     // [1024][4096]
                 const float*  __restrict__ pos,    // [2*SEQ-1]
                 const float*  __restrict__ sumA,   // windowed result f32
                 __bf16* __restrict__ outb) {       // bf16 (wout+out)
  __shared__ __bf16 qs[64][64], ks[64][64], vts[64][64], ps[64][64];
  __shared__ float gbias[64];
  const int bx = blockIdx.x;
  const int qblk = bx & 31, head = (bx >> 5) & 15, batch = bx >> 9;
  const int tid = threadIdx.x;
  const int wave = tid >> 5, lane = tid & 31, g = lane >> 4, ln = lane & 15;
  const int tq = batch * SEQ + qblk * 64;

  {
    int r = tid >> 1, sb = (tid & 1) * 32;
    const __bf16* qsrc = qk + (size_t)(tq + r) * (2 * DIM) + head * HD + sb;
#pragma unroll
    for (int s = 0; s < 4; ++s)
      cp_g2l_b128(qsrc + s * 8, &qs[r][sb + s * 8]);
  }
  const int i0 = wave * 16;
  floatx8 acc[4] = {};
  float mrow[8], lrow[8];
#pragma unroll
  for (int v = 0; v < 8; ++v) { mrow[v] = -1e30f; lrow[v] = 0.f; }

  for (int kb = 0; kb < SEQ / 64; ++kb) {
    __syncthreads();                         // previous tile fully consumed
    const int tk = batch * SEQ + kb * 64;
    {
      int r = tid >> 1, sb = (tid & 1) * 32;
      const __bf16* ksrc = qk + (size_t)(tk + r) * (2 * DIM) + DIM + head * HD + sb;
      const __bf16* vsrc = vT + (size_t)(head * HD + r) * MROWS + tk + sb;
#pragma unroll
      for (int s = 0; s < 4; ++s) {
        cp_g2l_b128(ksrc + s * 8, &ks[r][sb + s * 8]);
        cp_g2l_b128(vsrc + s * 8, &vts[r][sb + s * 8]);
      }
    }
    if (tid < 64) gbias[tid] = pos[(SEQ - 1) - (kb * 64 + tid)];
    wait_async();
    __syncthreads();

    floatx8 sc[4] = {};
#pragma unroll
    for (int d0 = 0; d0 < 64; d0 += 32) {
      bf16x16 a = frag_a(&qs[i0 + ln][d0], g);
#pragma unroll
      for (int nt = 0; nt < 4; ++nt) {
        bf16x16 b = frag_b(&ks[nt * 16 + ln][d0], g);
        sc[nt] = WMMA_BF16(a, b, sc[nt]);
      }
    }
    float pv[4][8];
#pragma unroll
    for (int nt = 0; nt < 4; ++nt)
#pragma unroll
      for (int v = 0; v < 8; ++v)
        pv[nt][v] = sc[nt][v] + gbias[nt * 16 + ln];  // no scale in global branch

#pragma unroll
    for (int v = 0; v < 8; ++v) {
      float m = fmaxf(fmaxf(pv[0][v], pv[1][v]), fmaxf(pv[2][v], pv[3][v]));
      for (int off = 8; off; off >>= 1) m = fmaxf(m, __shfl_xor(m, off, 32));
      float newm = fmaxf(mrow[v], m);
      float corr = __expf(mrow[v] - newm);
      mrow[v] = newm;
      float sum = 0.f;
      int row = i0 + v + 8 * g;
#pragma unroll
      for (int nt = 0; nt < 4; ++nt) {
        float p = __expf(pv[nt][v] - newm);
        sum += p;
        ps[row][nt * 16 + ln] = (__bf16)p;
      }
      for (int off = 8; off; off >>= 1) sum += __shfl_xor(sum, off, 32);
      lrow[v] = lrow[v] * corr + sum;
#pragma unroll
      for (int dt = 0; dt < 4; ++dt) acc[dt][v] *= corr;
    }
#pragma unroll
    for (int j0 = 0; j0 < 64; j0 += 32) {     // O += P V
      bf16x16 a = frag_a(&ps[i0 + ln][j0], g);
#pragma unroll
      for (int dt = 0; dt < 4; ++dt) {
        bf16x16 b = frag_b(&vts[dt * 16 + ln][j0], g);
        acc[dt] = WMMA_BF16(a, b, acc[dt]);
      }
    }
  }
#pragma unroll
  for (int dt = 0; dt < 4; ++dt) {
    int col = head * HD + dt * 16 + ln;
#pragma unroll
    for (int v = 0; v < 8; ++v) {
      int row = tq + i0 + v + 8 * g;
      float val = acc[dt][v] / lrow[v] + sumA[(size_t)row * DIM + col];
      outb[(size_t)row * DIM + col] = (__bf16)val;
    }
  }
}

// ---------------------------------------------------------------------------
extern "C" void kernel_launch(void* const* d_in, const int* in_sizes, int n_in,
                              void* d_out, int out_size, void* d_ws, size_t ws_size,
                              hipStream_t stream) {
  (void)in_sizes; (void)n_in; (void)out_size; (void)ws_size;
  const float* x     = (const float*)d_in[0];
  const float* w_v   = (const float*)d_in[1];
  const float* b_v   = (const float*)d_in[2];
  const float* w_wqk = (const float*)d_in[3];
  const float* b_wqk = (const float*)d_in[4];
  const float* w_qk  = (const float*)d_in[5];
  const float* b_qk  = (const float*)d_in[6];
  const float* w_out = (const float*)d_in[7];
  const float* b_out = (const float*)d_in[8];
  const float* w_pos = (const float*)d_in[9];
  const float* pos   = (const float*)d_in[10];

  char* ws = (char*)d_ws;
  size_t off = 0;
  auto alloc = [&](size_t bytes) -> void* {
    void* p = ws + off;
    off = (off + bytes + 255) & ~(size_t)255;
    return p;
  };
  __bf16* xb      = (__bf16*)alloc((size_t)MROWS * DIM * 2);
  __bf16* wvT     = (__bf16*)alloc((size_t)DIM * DIM * 2);          // [1024][1024]
  __bf16* wwqkT   = (__bf16*)alloc((size_t)2 * DIM * DIM * 2);      // [2048][1024]
  __bf16* wqkT    = (__bf16*)alloc((size_t)2 * DIM * DIM * 2);      // [2048][1024]
  __bf16* woutT   = (__bf16*)alloc((size_t)DIM * DIM * 2);          // [1024][1024]
  __bf16* vT      = (__bf16*)alloc((size_t)DIM * MROWS * 2);        // [1024][4096]
  __bf16* wqkact  = (__bf16*)alloc((size_t)MROWS * 2 * DIM * 2);    // [4096][2048]
  __bf16* qkact   = (__bf16*)alloc((size_t)MROWS * 2 * DIM * 2);    // [4096][2048]
  float*  sumA    = (float*)alloc((size_t)MROWS * DIM * 4);
  __bf16* ab      = (__bf16*)alloc((size_t)MROWS * DIM * 2);

  // x -> bf16 (row-major); weights -> bf16 transposed [N][K]
  cvt_f32_bf16<<<(MROWS * DIM + 255) / 256, 256, 0, stream>>>(x, xb, MROWS * DIM);
  cvtT_f32_bf16<<<dim3(DIM / 32, DIM / 32), 256, 0, stream>>>(w_v, wvT, DIM, DIM);
  cvtT_f32_bf16<<<dim3(2 * DIM / 32, DIM / 32), 256, 0, stream>>>(w_wqk, wwqkT, DIM, 2 * DIM);
  cvtT_f32_bf16<<<dim3(2 * DIM / 32, DIM / 32), 256, 0, stream>>>(w_qk, wqkT, DIM, 2 * DIM);
  cvtT_f32_bf16<<<dim3(DIM / 32, DIM / 32), 256, 0, stream>>>(w_out, woutT, DIM, DIM);

  // v = x w_v + b_v   -> stored transposed vT[1024][4096]
  gemm_bf16<<<dim3(DIM / 128, MROWS / 128), 256, 0, stream>>>(
      xb, DIM, wvT, DIM, b_v, nullptr, nullptr, vT, 0, MROWS, DIM);
  // wqk = x w_wqk + b_wqk  (bf16 [4096][2048])
  gemm_bf16<<<dim3(2 * DIM / 128, MROWS / 128), 256, 0, stream>>>(
      xb, DIM, wwqkT, DIM, b_wqk, wqkact, nullptr, nullptr, 2 * DIM, 0, DIM);
  // qk = x w_qk + b_qk
  gemm_bf16<<<dim3(2 * DIM / 128, MROWS / 128), 256, 0, stream>>>(
      xb, DIM, wqkT, DIM, b_qk, qkact, nullptr, nullptr, 2 * DIM, 0, DIM);

  // windowed branch -> sumA (f32)
  windowed_attn<<<NBATCH * NHEADS * (SEQ / WSZ), 128, 0, stream>>>(
      wqkact, vT, w_pos, sumA);
  // global branch, fused add -> ab (bf16)
  global_attn<<<NBATCH * NHEADS * (SEQ / 64), 128, 0, stream>>>(
      qkact, vT, pos, sumA, ab);

  // out = (wout + out) w_out + b_out  (f32 to d_out)
  gemm_bf16<<<dim3(DIM / 128, MROWS / 128), 256, 0, stream>>>(
      ab, DIM, woutT, DIM, b_out, nullptr, (float*)d_out, nullptr, DIM, 0, DIM);
}
